// HebbianTraceModule_51573967291054
// MI455X (gfx1250) — matrix operand order repaired and checked by reading, not compiled, not run on hardware
//
#include <hip/hip_runtime.h>

// ---------------------------------------------------------------------------
// HebbianTraceModule for MI455X (gfx1250, wave32, WMMA, async-to-LDS)
//
// Pipeline:
//   1) f32->bf16 weight converts (W_proj, W_val)
//   2) gather + LayerNorm -> tok_bf (raw) and ln_bf (normalized), bf16
//   3) Q = ln_bf @ W_proj^T      (WMMA bf16, async LDS staging, stored bf16)
//   4) V = tok_bf @ W_val^T      (same kernel)
//   5) upd[h] = Q_store^T V_store  (WMMA + global_atomic_add_f32)
//   6) trace = 0.99*trace0 + 0.1*upd/denom
//   7) Mmat[d, h*64+p] = sum_q trace[h,p,q] * W_out[d, h*64+q]  (tiny)
//   8) out = Q_addr @ Mmat^T     (WMMA, shifted-A rows, f32 out)
// ---------------------------------------------------------------------------

#define MTOT   32768      // B*S
#define SEQ    4096
#define DMODEL 768
#define HDIM   512        // N_HEADS * D_TRACE
#define NHEADS 8
#define DTRACE 64
#define KTOT   32752      // B*(S-2)

typedef __bf16 v16bf __attribute__((ext_vector_type(16)));
typedef __bf16 v8bf  __attribute__((ext_vector_type(8)));
typedef float  v8f   __attribute__((ext_vector_type(8)));

__device__ __forceinline__ unsigned short f2bf(float f) {
    unsigned u = __builtin_bit_cast(unsigned, f);
    unsigned r = u + 0x7fffu + ((u >> 16) & 1u);   // round to nearest even
    return (unsigned short)(r >> 16);
}

// ---- WMMA fragment loads from LDS (bf16, rows padded to 40 halfwords) -----
// A-matrix 16x32: lane L (m = L&15): lanes 0-15 hold K {0..7,16..23},
// lanes 16-31 hold K {8..15,24..31}; 2 bf16 packed per VGPR.
__device__ __forceinline__ v16bf load_frag_A(const unsigned short (*T)[40],
                                             int mbase, int lane) {
    int m = mbase + (lane & 15);
    int g = lane >> 4;
    v8bf lo = *reinterpret_cast<const v8bf*>(&T[m][g * 8]);
    v8bf hi = *reinterpret_cast<const v8bf*>(&T[m][16 + g * 8]);
    return __builtin_shufflevector(lo, hi, 0,1,2,3,4,5,6,7,8,9,10,11,12,13,14,15);
}
// B-matrix 32x16: lane L holds column n = L&15, K {0..7,16..23} (L<16) / {8..15,24..31}.
// LDS stores B transposed as [n][k] so each lane reads two contiguous runs of 8 bf16.
__device__ __forceinline__ v16bf load_frag_B(const unsigned short (*T)[40],
                                             int nbase, int lane) {
    int n  = nbase + (lane & 15);
    int kb = (lane >> 4) * 8;
    v8bf lo = *reinterpret_cast<const v8bf*>(&T[n][kb]);
    v8bf hi = *reinterpret_cast<const v8bf*>(&T[n][kb + 16]);
    return __builtin_shufflevector(lo, hi, 0,1,2,3,4,5,6,7,8,9,10,11,12,13,14,15);
}

// 16B global -> LDS, no VGPR round-trip; tracked by ASYNCcnt.
__device__ __forceinline__ void async_copy16(unsigned lds_addr, const void* gptr) {
    unsigned long long ga = (unsigned long long)(uintptr_t)gptr;
    asm volatile("global_load_async_to_lds_b128 %0, %1, off"
                 :: "v"(lds_addr), "v"(ga) : "memory");
}
__device__ __forceinline__ void wait_async0() {
    asm volatile("s_wait_asynccnt 0x0" ::: "memory");
}

// ---------------------------------------------------------------------------
__global__ void f32_to_bf16_kernel(const float* __restrict__ s,
                                   unsigned short* __restrict__ d, int n) {
    int i = blockIdx.x * 256 + threadIdx.x;
    if (i < n) d[i] = f2bf(s[i]);
}

__global__ void zero_f32_kernel(float* __restrict__ p, int n) {
    int i = blockIdx.x * 256 + threadIdx.x;
    if (i < n) p[i] = 0.0f;
}

// ---------------------------------------------------------------------------
// gather + LayerNorm; one block (256 threads) per token row of 768 floats
__global__ __launch_bounds__(256)
void gather_ln_kernel(const int* __restrict__ ids, const float* __restrict__ wte,
                      const float* __restrict__ lnw, const float* __restrict__ lnb,
                      unsigned short* __restrict__ tokbf,
                      unsigned short* __restrict__ lnbf) {
    __shared__ float s1[256], s2[256];
    const int row = blockIdx.x;
    const int tid = threadIdx.x;
    const int id  = ids[row];
    const float* src = wte + (size_t)id * DMODEL;
    float x0 = src[tid], x1 = src[tid + 256], x2 = src[tid + 512];
    s1[tid] = x0 + x1 + x2;
    s2[tid] = x0 * x0 + x1 * x1 + x2 * x2;
    __syncthreads();
    for (int off = 128; off > 0; off >>= 1) {
        if (tid < off) { s1[tid] += s1[tid + off]; s2[tid] += s2[tid + off]; }
        __syncthreads();
    }
    const float mean = s1[0] * (1.0f / DMODEL);
    const float var  = s2[0] * (1.0f / DMODEL) - mean * mean;
    const float rstd = rsqrtf(var + 1e-5f);
    const size_t o = (size_t)row * DMODEL;
    tokbf[o + tid]       = f2bf(x0);
    tokbf[o + tid + 256] = f2bf(x1);
    tokbf[o + tid + 512] = f2bf(x2);
    lnbf[o + tid]       = f2bf((x0 - mean) * rstd * lnw[tid]       + lnb[tid]);
    lnbf[o + tid + 256] = f2bf((x1 - mean) * rstd * lnw[tid + 256] + lnb[tid + 256]);
    lnbf[o + tid + 512] = f2bf((x2 - mean) * rstd * lnw[tid + 512] + lnb[tid + 512]);
}

// ---------------------------------------------------------------------------
// C(MxN) = A(MxK) * B(NxK)^T ; bf16 inputs, f32 accumulate via WMMA.
// 128x128 block tile, BK=32, 256 threads = 8 waves (wm 0..3 x wn 0..1),
// each wave owns a 32x64 C slab -> 8 WMMA per K-step.
// Double-buffered LDS fed by global_load_async_to_lds_b128 (ASYNCcnt),
// one barrier per K-step. sched_barrier(0) pins all 12 ds_load_b128 ahead of
// the 8 WMMAs so they drain behind a single dscnt wait and the WMMAs issue
// as one burst. aShift: A row m reads row m-1 (zeros at seq starts)==Q_addr.
__global__ __launch_bounds__(256)
void gemm_bf16_nt(const unsigned short* __restrict__ A, int lda,
                  const unsigned short* __restrict__ Bw, int ldb,
                  void* __restrict__ C, int ldc,
                  int K, int outBF16, int aShift) {
    __shared__ alignas(16) unsigned short At[2][128][40];
    __shared__ alignas(16) unsigned short Bt[2][128][40];
    const int tid = threadIdx.x, lane = tid & 31, wid = tid >> 5;
    const int wm = wid & 3, wn = wid >> 2;
    const int m0 = blockIdx.y * 128, n0 = blockIdx.x * 128;

    v8f acc[2][4];
#pragma unroll
    for (int a = 0; a < 2; ++a)
#pragma unroll
        for (int b = 0; b < 4; ++b) acc[a][b] = v8f{};

    // K-invariant staging descriptors: each thread owns 2 chunks of 16B
    // in both tiles.
    int  aRow[2];          // global A row feeding this chunk
    bool aZero[2];         // Q_addr zero row (sequence start)
    int  ldsOff[2];        // element offset of chunk inside a [128][40] tile
    int  segOf[2];
#pragma unroll
    for (int c = 0; c < 2; ++c) {
        const int chunk = tid + c * 256;     // 0..511
        const int r = chunk >> 2;            // 0..127
        const int seg = chunk & 3;           // 8 bf16 each
        ldsOff[c] = r * 40 + seg * 8;
        segOf[c]  = seg;
        const int m = m0 + r;
        aRow[c] = m; aZero[c] = false;
        if (aShift) {
            if ((m & (SEQ - 1)) == 0) aZero[c] = true; else aRow[c] = m - 1;
        }
    }

    auto stage = [&](int buf, int k0) {
#pragma unroll
        for (int c = 0; c < 2; ++c) {
            const int r = (tid + c * 256) >> 2;
            unsigned aLds = (unsigned)(uintptr_t)&At[buf][0][0] + (unsigned)(ldsOff[c] * 2);
            unsigned bLds = (unsigned)(uintptr_t)&Bt[buf][0][0] + (unsigned)(ldsOff[c] * 2);
            if (!aZero[c]) {
                async_copy16(aLds, &A[(size_t)aRow[c] * lda + k0 + segOf[c] * 8]);
            } else {
                uint4 z = {0, 0, 0, 0};
                *reinterpret_cast<uint4*>(&At[buf][0][0] + ldsOff[c]) = z;
            }
            async_copy16(bLds, &Bw[(size_t)(n0 + r) * ldb + k0 + segOf[c] * 8]);
        }
    };

    const int nsteps = K >> 5;
    stage(0, 0);
    wait_async0();
    __syncthreads();

    for (int s = 0; s < nsteps; ++s) {
        const int cur = s & 1;
        if (s + 1 < nsteps) stage(cur ^ 1, (s + 1) * 32);   // overlap with WMMA

        // Preload ALL fragments first (12 ds_load_b128 in flight) ...
        v16bf af0 = load_frag_A(At[cur], wm * 32,      lane);
        v16bf af1 = load_frag_A(At[cur], wm * 32 + 16, lane);
        v16bf bfr[4];
#pragma unroll
        for (int tn = 0; tn < 4; ++tn)
            bfr[tn] = load_frag_B(Bt[cur], wn * 64 + tn * 16, lane);
        // ... then forbid the scheduler from sinking loads into the WMMA
        // burst: one dscnt drain, 8 back-to-back WMMAs.
        __builtin_amdgcn_sched_barrier(0);
#pragma unroll
        for (int tn = 0; tn < 4; ++tn) {
            acc[0][tn] = __builtin_amdgcn_wmma_f32_16x16x32_bf16(
                false, af0, false, bfr[tn], (short)0, acc[0][tn], false, false);
            acc[1][tn] = __builtin_amdgcn_wmma_f32_16x16x32_bf16(
                false, af1, false, bfr[tn], (short)0, acc[1][tn], false, false);
        }
        wait_async0();       // next buffer fully in LDS
        __syncthreads();     // all waves done reading cur; safe to overwrite
    }

    // C layout per 16x16 tile: lane = n&15 (+16 for m>=8), vgpr v = m&7
    const int rhalf = (lane >> 4) ? 8 : 0;
    const int cl    = lane & 15;
#pragma unroll
    for (int tm = 0; tm < 2; ++tm) {
        const int rbase = m0 + wm * 32 + tm * 16 + rhalf;
#pragma unroll
        for (int tn = 0; tn < 4; ++tn) {
            const int c0 = n0 + wn * 64 + tn * 16 + cl;
            for (int v = 0; v < 8; ++v) {
                const int r = rbase + v;
                if (outBF16) {
                    ((unsigned short*)C)[(size_t)r * ldc + c0] = f2bf(acc[tm][tn][v]);
                } else {
                    ((float*)C)[(size_t)r * ldc + c0] = acc[tm][tn][v];
                }
            }
        }
    }
}

// ---------------------------------------------------------------------------
// upd[h,p,q] += sum_i Q[i,p] * V[i+2,q]  over a K-chunk of 512 positions.
// Grid: (64 k-chunks, 8 heads). Both tiles staged transposed ([p][i]/[q][i])
// so WMMA fragments read contiguous K runs. Reduction via f32 global atomics.
__global__ __launch_bounds__(256)
void trace_accum_kernel(const unsigned short* __restrict__ Qbf,
                        const unsigned short* __restrict__ Vbf,
                        float* __restrict__ acc) {
    __shared__ alignas(16) unsigned short QT[64][40];
    __shared__ alignas(16) unsigned short VT[64][40];
    const int tid = threadIdx.x, lane = tid & 31, wid = tid >> 5;
    const int wm = wid & 3, wn = wid >> 2;
    const int h    = blockIdx.y;
    const int base = blockIdx.x * 512;
    const int ii = tid >> 3;            // 0..31 (k position in tile)
    const int p0 = (tid & 7) * 8;       // 0..56 (8 features per thread)
    v8f acc0 = {}; v8f acc1 = {};

    for (int step = 0; step < 16; ++step) {
        const int i_flat = base + step * 32 + ii;
        uint4 qv = {0, 0, 0, 0}, vv = {0, 0, 0, 0};
        if (i_flat < KTOT) {
            const int b = i_flat / (SEQ - 2);
            const int i = i_flat - b * (SEQ - 2);
            const size_t qoff = ((size_t)(b * SEQ + i))     * HDIM + h * 64 + p0;
            const size_t voff = ((size_t)(b * SEQ + i + 2)) * HDIM + h * 64 + p0;
            qv = *reinterpret_cast<const uint4*>(&Qbf[qoff]);
            vv = *reinterpret_cast<const uint4*>(&Vbf[voff]);
        }
        union U { uint4 u; unsigned short s[8]; } uq, uv;
        uq.u = qv; uv.u = vv;
#pragma unroll
        for (int j = 0; j < 8; ++j) {
            QT[p0 + j][ii] = uq.s[j];
            VT[p0 + j][ii] = uv.s[j];
        }
        __syncthreads();
        // preload fragments, then fire both WMMAs as a burst
        v16bf af = load_frag_A(QT, wm * 16, lane);
        v16bf b0 = load_frag_B(VT, wn * 32, lane);
        v16bf b1 = load_frag_B(VT, wn * 32 + 16, lane);
        __builtin_amdgcn_sched_barrier(0);
        acc0 = __builtin_amdgcn_wmma_f32_16x16x32_bf16(false, af, false, b0,
                                                       (short)0, acc0, false, false);
        acc1 = __builtin_amdgcn_wmma_f32_16x16x32_bf16(false, af, false, b1,
                                                       (short)0, acc1, false, false);
        __syncthreads();
    }
    const int rbase = wm * 16 + ((lane >> 4) ? 8 : 0);
    const int c0    = wn * 32 + (lane & 15);
    float* ah = acc + h * (DTRACE * DTRACE);
    for (int v = 0; v < 8; ++v) {
        atomicAdd(&ah[(rbase + v) * DTRACE + c0],      acc0[v]);
        atomicAdd(&ah[(rbase + v) * DTRACE + c0 + 16], acc1[v]);
    }
}

__global__ void trace_final_kernel(const float* __restrict__ acc,
                                   const float* __restrict__ trace0,
                                   float* __restrict__ trace) {
    int i = blockIdx.x * 256 + threadIdx.x;
    if (i < NHEADS * DTRACE * DTRACE)
        trace[i] = 0.99f * trace0[i] + (0.1f / (float)KTOT) * acc[i];
}

// Mmat[d, h*64+p] = sum_q trace[h,p,q] * W_out[d, h*64+q]   (768x512 bf16)
__global__ void combine_wout_kernel(const float* __restrict__ trace,
                                    const float* __restrict__ Wout,
                                    unsigned short* __restrict__ Mmat) {
    int idx = blockIdx.x * 256 + threadIdx.x;
    if (idx >= DMODEL * HDIM) return;
    const int d = idx >> 9;          // / 512
    const int c = idx & 511;
    const int h = c >> 6, p = c & 63;
    const float* tr = trace + h * (DTRACE * DTRACE) + p * DTRACE;
    const float* wo = Wout + (size_t)d * HDIM + h * DTRACE;
    float s = 0.0f;
#pragma unroll 8
    for (int q = 0; q < DTRACE; ++q) s += tr[q] * wo[q];
    Mmat[idx] = f2bf(s);
}

// ---------------------------------------------------------------------------
extern "C" void kernel_launch(void* const* d_in, const int* in_sizes, int n_in,
                              void* d_out, int out_size, void* d_ws, size_t ws_size,
                              hipStream_t stream) {
    const int*   ids    = (const int*)d_in[0];
    const float* wte    = (const float*)d_in[1];
    const float* Wproj  = (const float*)d_in[2];
    const float* lnw    = (const float*)d_in[3];
    const float* lnb    = (const float*)d_in[4];
    const float* Wval   = (const float*)d_in[5];
    const float* Wout   = (const float*)d_in[6];
    const float* trace0 = (const float*)d_in[7];

    char* ws = (char*)d_ws;
    size_t off = 0;
    auto alloc = [&](size_t bytes) -> void* {
        void* p = ws + off;
        off += (bytes + 255) & ~(size_t)255;
        return p;
    };
    unsigned short* tokbf   = (unsigned short*)alloc((size_t)MTOT * DMODEL * 2);
    unsigned short* lnbf    = (unsigned short*)alloc((size_t)MTOT * DMODEL * 2);
    unsigned short* wprojbf = (unsigned short*)alloc((size_t)HDIM * DMODEL * 2);
    unsigned short* wvalbf  = (unsigned short*)alloc((size_t)HDIM * DMODEL * 2);
    unsigned short* Qbf     = (unsigned short*)alloc((size_t)MTOT * HDIM * 2);
    unsigned short* Vbf     = (unsigned short*)alloc((size_t)MTOT * HDIM * 2);
    unsigned short* Mmat    = (unsigned short*)alloc((size_t)DMODEL * HDIM * 2);
    float*          acc     = (float*)alloc((size_t)NHEADS * DTRACE * DTRACE * 4);
    float*          trace   = (float*)alloc((size_t)NHEADS * DTRACE * DTRACE * 4);

    const int nW = HDIM * DMODEL;                      // 393216
    f32_to_bf16_kernel<<<(nW + 255) / 256, 256, 0, stream>>>(Wproj, wprojbf, nW);
    f32_to_bf16_kernel<<<(nW + 255) / 256, 256, 0, stream>>>(Wval,  wvalbf,  nW);

    gather_ln_kernel<<<MTOT, 256, 0, stream>>>(ids, wte, lnw, lnb, tokbf, lnbf);

    dim3 gQV(HDIM / 128, MTOT / 128);                  // (4, 256)
    gemm_bf16_nt<<<gQV, 256, 0, stream>>>(lnbf,  DMODEL, wprojbf, DMODEL,
                                          (void*)Qbf, HDIM, DMODEL, 1, 0);
    gemm_bf16_nt<<<gQV, 256, 0, stream>>>(tokbf, DMODEL, wvalbf,  DMODEL,
                                          (void*)Vbf, HDIM, DMODEL, 1, 0);

    const int nT = NHEADS * DTRACE * DTRACE;           // 32768
    zero_f32_kernel<<<(nT + 255) / 256, 256, 0, stream>>>(acc, nT);

    dim3 gT(64, NHEADS);
    trace_accum_kernel<<<gT, 256, 0, stream>>>(Qbf, Vbf, acc);
    trace_final_kernel<<<(nT + 255) / 256, 256, 0, stream>>>(acc, trace0, trace);

    combine_wout_kernel<<<(DMODEL * HDIM + 255) / 256, 256, 0, stream>>>(trace, Wout, Mmat);

    dim3 gOut(DMODEL / 128, MTOT / 128);               // (6, 256)
    gemm_bf16_nt<<<gOut, 256, 0, stream>>>(Qbf, HDIM, Mmat, HDIM,
                                           d_out, DMODEL, HDIM, 0, 1);
}
